// SurfaceGraphCommunication_738734374947
// MI455X (gfx1250) — matrix operand order repaired and checked by reading, not compiled, not run on hardware
//
#include <hip/hip_runtime.h>

// SurfaceGraphCommunication on gfx1250 (MI455X), wave32 + WMMA f16 + async LDS.
//
// Pipeline (all on `stream`, dependencies via stream order):
//   1. k_prep      : transpose+convert 4 weight matrices to f16 [E][K]
//   2. k_pre  (x2) : h = relu(x @ W + b), store h(f16 row-major) and hT(f16 [B][D][N])
//   3. k_msg_s     : xs_msg = (wu @ xg_h) / (rowsum+eps); wu recomputed from coords
//                    per A-fragment; B tiles double-buffered in LDS via async loads
//   4. k_msg_g     : xg_msg = wu^T/rowsum @ xs_h; B tiles + coords + rowsum
//                    double-buffered in LDS via global_load_async_to_lds_b128
//   5. k_post (x2) : out = relu(concat(h, msg) @ Wpost + b), K=256
//
// Dims fixed by the reference: B=4, Ns=16384, Ng=2048, D=128.

#define DEV __device__ __forceinline__

typedef _Float16 v16h __attribute__((ext_vector_type(16)));
typedef _Float16 v8h  __attribute__((ext_vector_type(8)));
typedef float    v8f  __attribute__((ext_vector_type(8)));
typedef float    v4f  __attribute__((ext_vector_type(4)));

union V16H { v16h v; v8h h[2]; };

// CDNA5 async global->LDS copy (per-lane addresses), tracked by ASYNCcnt.
#define ASYNC_LOAD_B128(la, ga)                                               \
  asm volatile("global_load_async_to_lds_b128 %0, %1, off" ::"v"(la), "v"(ga) \
               : "memory")
#define WAIT_ASYNC0() asm volatile("s_wait_asynccnt 0x0" ::: "memory")

DEV v8f wmma_f16(v16h a, v16h b, v8f c) {
  // D = A(16x32 f16) * B(32x16 f16) + C(16x16 f32)
  return __builtin_amdgcn_wmma_f32_16x16x32_f16(false, a, false, b, (short)0, c,
                                                false, false);
}

// A-fragment (16x32 f16) element e on lane L maps to row M=L&15,
// K = (e&7) + ((e>>3)<<4) + ((L>>4)<<3)  -> two contiguous runs of 8.
DEV v16h load_a_f32(const float* __restrict__ row, int kb, int hf) {
  const float* p0 = row + kb + hf * 8;
  v4f a0 = *(const v4f*)(p0);
  v4f a1 = *(const v4f*)(p0 + 4);
  v4f b0 = *(const v4f*)(p0 + 16);
  v4f b1 = *(const v4f*)(p0 + 20);
  V16H r;
#pragma unroll
  for (int i = 0; i < 4; ++i) {
    r.v[i]      = (_Float16)a0[i];
    r.v[4 + i]  = (_Float16)a1[i];
    r.v[8 + i]  = (_Float16)b0[i];
    r.v[12 + i] = (_Float16)b1[i];
  }
  return r.v;
}

DEV v16h load_a_f16(const _Float16* __restrict__ row, int kb, int hf) {
  V16H r;
  r.h[0] = *(const v8h*)(row + kb + hf * 8);
  r.h[1] = *(const v8h*)(row + kb + hf * 8 + 16);
  return r.v;
}

// B-fragment (32x16 f16): lanes 0-15 hold K 0..7,16..23 of column n=lane,
// lanes 16-31 hold K 8..15,24..31.  `col` points at a transposed [n][K] row.
DEV v16h load_b_col(const _Float16* __restrict__ col, int kb, int hf) {
  V16H r;
  const _Float16* p = col + kb + hf * 8;
  r.h[0] = *(const v8h*)(p);
  r.h[1] = *(const v8h*)(p + 16);
  return r.v;
}

// B-fragment from an LDS-staged [d=0..127][k=0..31] f16 tile.
DEV v16h load_b_lds(const _Float16* tile, int nc, int hf, int j) {
  V16H r;
  const _Float16* p = tile + (j * 16 + nc) * 32 + hf * 8;
  r.h[0] = *(const v8h*)(p);
  r.h[1] = *(const v8h*)(p + 16);
  return r.v;
}

// -------------------------------------------------------------------------
// 1. Weight prep: W[K][E] f32 -> WT[E][K] f16
__global__ void k_prep(const float* __restrict__ Ws_pre, const float* __restrict__ Wg_pre,
                       const float* __restrict__ Ws_post, const float* __restrict__ Wg_post,
                       _Float16* __restrict__ WsPreT, _Float16* __restrict__ WgPreT,
                       _Float16* __restrict__ WsPostT, _Float16* __restrict__ WgPostT) {
  int t = blockIdx.x * blockDim.x + threadIdx.x;
  if (t < 16384) {
    int k = t >> 7, e = t & 127;
    WsPreT[e * 128 + k] = (_Float16)Ws_pre[t];
  } else if (t < 32768) {
    int u = t - 16384; int k = u >> 7, e = u & 127;
    WgPreT[e * 128 + k] = (_Float16)Wg_pre[u];
  } else if (t < 65536) {
    int u = t - 32768; int k = u >> 7, e = u & 127;   // Ws_post: [256][128]
    WsPostT[e * 256 + k] = (_Float16)Ws_post[u];
  } else if (t < 98304) {
    int u = t - 65536; int k = u >> 7, e = u & 127;
    WgPostT[e * 256 + k] = (_Float16)Wg_post[u];
  }
}

// -------------------------------------------------------------------------
// 2. Pre-encoder: h = relu(x @ W + b).  One wave = 16 rows x 128 cols.
__global__ void k_pre(const float* __restrict__ x, const _Float16* __restrict__ WT,
                      const float* __restrict__ bias,
                      _Float16* __restrict__ h16, _Float16* __restrict__ hT16,
                      int Nb) {
  const int wave = (blockIdx.x * blockDim.x + threadIdx.x) >> 5;
  const int lane = threadIdx.x & 31;
  const int hf = lane >> 4, nc = lane & 15;
  const int rowbase = wave * 16;
  const int b = rowbase / Nb, n0 = rowbase % Nb;
  const float* arow = x + (size_t)(rowbase + nc) * 128;

  v8f c[8] = {};
#pragma unroll
  for (int kc = 0; kc < 4; ++kc) {
    const int kb = kc * 32;
    v16h a = load_a_f32(arow, kb, hf);
#pragma unroll
    for (int j = 0; j < 8; ++j) {
      v16h bf = load_b_col(WT + (size_t)(j * 16 + nc) * 128, kb, hf);
      c[j] = wmma_f16(a, bf, c[j]);
    }
  }
  const int Mb = hf * 8;
#pragma unroll
  for (int j = 0; j < 8; ++j) {
    const int e = j * 16 + nc;
    const float bv = bias[e];
    v8h hh;
#pragma unroll
    for (int r = 0; r < 8; ++r) {
      float v = c[j][r] + bv;
      v = v > 0.f ? v : 0.f;
      hh[r] = (_Float16)v;
      h16[(size_t)(rowbase + Mb + r) * 128 + e] = hh[r];
    }
    // hT: [b][e][n] -> 8 consecutive n per lane = one 16B store
    *(v8h*)(hT16 + ((size_t)b * 128 + e) * Nb + (n0 + Mb)) = hh;
  }
}

// -------------------------------------------------------------------------
// 3. Surface messages: xs_msg = (wu @ xg_h) / (rowsum + 1e-8); writes rowsum.
//    Block = 4 waves = 64 surface rows.  vg coords: one-time 32 KB LDS preload.
//    Per 32-wide m-chunk the 8 KB xg_hT tile is double-buffered in LDS via
//    block-partitioned async loads (512 x 16B granules / 128 threads).
__global__ void __launch_bounds__(128) k_msg_s(
    const float* __restrict__ vs, const float* __restrict__ vg,
    const _Float16* __restrict__ xg_hT,  // [B][128][Ng]
    _Float16* __restrict__ msg16, float* __restrict__ rowsum,
    int Ns, int Ng) {
  __shared__ v4f svg[2048];
  __shared__ alignas(16) _Float16 sB[2][128 * 32];

  const int tid = threadIdx.x;
  const int b = (blockIdx.x * 64) / Ns;
  for (int m = tid; m < Ng; m += blockDim.x) {
    const float* p = vg + ((size_t)b * Ng + m) * 3;
    v4f q = {p[0], p[1], p[2], 0.f};
    svg[m] = q;
  }
  __syncthreads();

  const int wv = tid >> 5;
  const int lane = tid & 31;
  const int hf = lane >> 4, nc = lane & 15;
  const int rowg = blockIdx.x * 64 + wv * 16;   // global surface row
  const int n0 = rowg % Ns;
  const float* vsp = vs + ((size_t)b * Ns + n0 + nc) * 3;
  const float sx = vsp[0], sy = vsp[1], sz = vsp[2];
  const _Float16* hTb = xg_hT + (size_t)b * 128 * Ng;

  auto issueB = [&](int buf, int mu) {
#pragma unroll
    for (int it = 0; it < 4; ++it) {
      const int g = tid + 128 * it;
      const int d = g >> 2, part = g & 3;
      const unsigned long long ga =
          (unsigned long long)(size_t)(hTb + (size_t)d * Ng + mu + part * 8);
      const unsigned la = (unsigned)(size_t)(&sB[buf][d * 32 + part * 8]);
      ASYNC_LOAD_B128(la, ga);
    }
  };
  issueB(0, 0);   // prologue: chunk 0 in flight

  v8f c[8] = {};
  float rs = 0.f;
  const int NCH = Ng / 32;
  for (int i = 0; i < NCH; ++i) {
    const int cur = i & 1, nxt = cur ^ 1;
    const int mc = i * 32;
    WAIT_ASYNC0();        // own async loads for chunk i landed in LDS
    __syncthreads();      // everyone's chunk i landed; prior reads of nxt done
    if (i + 1 < NCH) issueB(nxt, mc + 32);

    V16H a;
#pragma unroll
    for (int e = 0; e < 16; ++e) {
      const int k = (e & 7) + ((e >> 3) << 4) + (hf << 3);
      v4f g = svg[mc + k];
      float dx = sx - g.x, dy = sy - g.y, dz = sz - g.z;
      float d2 = dx * dx + dy * dy + dz * dz;
      // exp(-d2/sigma^2) = exp2(-d2 * log2(e)/6.25), cutoff d2 < 64
      float w = d2 < 64.f ? __builtin_amdgcn_exp2f(d2 * -0.23083121f) : 0.f;
      rs += w;
      a.v[e] = (_Float16)w;
    }
#pragma unroll
    for (int j = 0; j < 8; ++j)
      c[j] = wmma_f16(a.v, load_b_lds(&sB[cur][0], nc, hf, j), c[j]);
    // Buffer written at iter i is read at i+1 and next overwritten only after
    // the barrier at the top of iter i+2 -> one barrier per iteration suffices.
  }

  // lane L and L+16 cover disjoint K halves of row M=L&15
  rs += __shfl_xor(rs, 16, 32);
  if (lane < 16) rowsum[(size_t)b * Ns + n0 + lane] = rs;

  const int Mb = hf * 8;
  float inv[8];
#pragma unroll
  for (int r = 0; r < 8; ++r)
    inv[r] = 1.f / (__shfl(rs, Mb + r, 32) + 1e-8f);
#pragma unroll
  for (int j = 0; j < 8; ++j) {
    const int e = j * 16 + nc;
#pragma unroll
    for (int r = 0; r < 8; ++r)
      msg16[(size_t)(rowg + Mb + r) * 128 + e] = (_Float16)(c[j][r] * inv[r]);
  }
}

// -------------------------------------------------------------------------
// 4. Graph messages: xg_msg[m] = sum_n wu[n,m]/(rowsum[n]+eps) * xs_h[n,:]
//    Block = 4 waves = 64 graph rows sharing one n-chunk stream.
//    Per 32-wide n-chunk, LDS receives (double-buffered, async):
//      sB[buf]  : xs_hT tile, [d=0..127][k=0..31] f16      (8 KB)
//      sAux[buf]: 96 f32 raw vs coords + 32 f32 rowsum     (512 B)
__global__ void __launch_bounds__(128) k_msg_g(
    const float* __restrict__ vs, const float* __restrict__ vg,
    const _Float16* __restrict__ xs_hT,  // [B][128][Ns]
    const float* __restrict__ rowsum,
    _Float16* __restrict__ msg16, int Ns, int Ng) {
  __shared__ alignas(16) _Float16 sB[2][128 * 32];
  __shared__ alignas(16) float sAux[2][128];

  const int tid = threadIdx.x;
  const int wv = tid >> 5, lane = tid & 31;
  const int hf = lane >> 4, nc = lane & 15;
  const int rowg = blockIdx.x * 64 + wv * 16;   // global graph row
  const int b = rowg / Ng, m0 = rowg % Ng;
  const float* vgp = vg + ((size_t)b * Ng + m0 + nc) * 3;
  const float gx = vgp[0], gy = vgp[1], gz = vgp[2];
  const _Float16* hTb = xs_hT + (size_t)b * 128 * Ns;
  const float* rsb = rowsum + (size_t)b * Ns;
  const float* vsb = vs + (size_t)b * Ns * 3;

  // Block-wide partition of the 8 KB B tile: 512 x 16B granules / 128 threads.
  auto issueB = [&](int buf, int nu) {
#pragma unroll
    for (int it = 0; it < 4; ++it) {
      const int g = tid + 128 * it;
      const int d = g >> 2, part = g & 3;
      const unsigned long long ga =
          (unsigned long long)(size_t)(hTb + (size_t)d * Ns + nu + part * 8);
      const unsigned la = (unsigned)(size_t)(&sB[buf][d * 32 + part * 8]);
      ASYNC_LOAD_B128(la, ga);
    }
  };
  // One wave-wide instruction: lanes 0-23 fetch 96 f32 of raw vs coords,
  // lanes 24-31 fetch the 32 f32 rowsum values.
  auto issueAux = [&](int buf, int nu) {
    const float* gsrc = (lane < 24) ? (vsb + (size_t)nu * 3 + lane * 4)
                                    : (rsb + nu + (lane - 24) * 4);
    const unsigned long long ga = (unsigned long long)(size_t)gsrc;
    const unsigned la = (unsigned)((size_t)&sAux[buf][0] + lane * 16);
    ASYNC_LOAD_B128(la, ga);
  };

  // Prologue: chunk 0 in flight.
  issueB(0, 0);
  if (wv == 0) issueAux(0, 0);

  v8f c[8] = {};
  const int NCH = Ns / 32;
  for (int i = 0; i < NCH; ++i) {
    const int cur = i & 1, nxt = cur ^ 1;
    WAIT_ASYNC0();        // own async loads for chunk i landed in LDS
    __syncthreads();      // everyone's chunk i landed; prior reads of nxt done
    if (i + 1 < NCH) {
      issueB(nxt, (i + 1) * 32);
      if (wv == 0) issueAux(nxt, (i + 1) * 32);
    }
    V16H a;
#pragma unroll
    for (int e = 0; e < 16; ++e) {
      const int k = (e & 7) + ((e >> 3) << 4) + (hf << 3);
      const float px = sAux[cur][k * 3 + 0];
      const float py = sAux[cur][k * 3 + 1];
      const float pz = sAux[cur][k * 3 + 2];
      float dx = gx - px, dy = gy - py, dz = gz - pz;
      float d2 = dx * dx + dy * dy + dz * dz;
      float w = d2 < 64.f ? __builtin_amdgcn_exp2f(d2 * -0.23083121f) : 0.f;
      w *= __builtin_amdgcn_rcpf(sAux[cur][96 + k] + 1e-8f);
      a.v[e] = (_Float16)w;
    }
#pragma unroll
    for (int j = 0; j < 8; ++j)
      c[j] = wmma_f16(a.v, load_b_lds(&sB[cur][0], nc, hf, j), c[j]);
  }

  const int Mb = hf * 8;
#pragma unroll
  for (int j = 0; j < 8; ++j) {
    const int e = j * 16 + nc;
#pragma unroll
    for (int r = 0; r < 8; ++r)
      msg16[(size_t)(rowg + Mb + r) * 128 + e] = (_Float16)c[j][r];
  }
}

// -------------------------------------------------------------------------
// 5. Post block: out = relu(concat(h, msg) @ Wpost + b), K = 256
__global__ void k_post(const _Float16* __restrict__ h16, const _Float16* __restrict__ msg16,
                       const _Float16* __restrict__ WT,   // [128][256]
                       const float* __restrict__ bias,
                       float* __restrict__ out, int Nb, int node_off, int node_stride) {
  const int wave = (blockIdx.x * blockDim.x + threadIdx.x) >> 5;
  const int lane = threadIdx.x & 31;
  const int hf = lane >> 4, nc = lane & 15;
  const int rowbase = wave * 16;
  const int b = rowbase / Nb, n0 = rowbase % Nb;
  const _Float16* arow_h = h16 + (size_t)(rowbase + nc) * 128;
  const _Float16* arow_m = msg16 + (size_t)(rowbase + nc) * 128;

  v8f c[8] = {};
#pragma unroll
  for (int kc = 0; kc < 8; ++kc) {
    const int kb = kc * 32;
    v16h a = (kc < 4) ? load_a_f16(arow_h, kb, hf)
                      : load_a_f16(arow_m, kb - 128, hf);
#pragma unroll
    for (int j = 0; j < 8; ++j) {
      v16h bf = load_b_col(WT + (size_t)(j * 16 + nc) * 256, kb, hf);
      c[j] = wmma_f16(a, bf, c[j]);
    }
  }
  const int Mb = hf * 8;
  const size_t outrow = (size_t)b * node_stride + node_off + n0;
#pragma unroll
  for (int j = 0; j < 8; ++j) {
    const int e = j * 16 + nc;
    const float bv = bias[e];
#pragma unroll
    for (int r = 0; r < 8; ++r) {
      float v = c[j][r] + bv;
      out[(outrow + Mb + r) * 128 + e] = v > 0.f ? v : 0.f;
    }
  }
}

// -------------------------------------------------------------------------
extern "C" void kernel_launch(void* const* d_in, const int* in_sizes, int n_in,
                              void* d_out, int out_size, void* d_ws, size_t ws_size,
                              hipStream_t stream) {
  (void)in_sizes; (void)n_in; (void)out_size; (void)ws_size;
  constexpr int B = 4, Ns = 16384, Ng = 2048, D = 128;

  const float* xs      = (const float*)d_in[0];
  const float* xg      = (const float*)d_in[1];
  const float* vs      = (const float*)d_in[2];
  const float* vg      = (const float*)d_in[3];
  const float* Ws_pre  = (const float*)d_in[4];
  const float* bs_pre  = (const float*)d_in[5];
  const float* Wg_pre  = (const float*)d_in[6];
  const float* bg_pre  = (const float*)d_in[7];
  const float* Ws_post = (const float*)d_in[8];
  const float* bs_post = (const float*)d_in[9];
  const float* Wg_post = (const float*)d_in[10];
  const float* bg_post = (const float*)d_in[11];
  float* out = (float*)d_out;

  char* ws = (char*)d_ws;
  size_t off = 0;
  auto carve = [&](size_t bytes) -> char* {
    char* p = ws + off;
    off += (bytes + 255) & ~(size_t)255;
    return p;
  };
  _Float16* WsPreT  = (_Float16*)carve((size_t)D * D * 2);
  _Float16* WgPreT  = (_Float16*)carve((size_t)D * D * 2);
  _Float16* WsPostT = (_Float16*)carve((size_t)2 * D * D * 2);
  _Float16* WgPostT = (_Float16*)carve((size_t)2 * D * D * 2);
  _Float16* xs_h    = (_Float16*)carve((size_t)B * Ns * D * 2);
  _Float16* xs_hT   = (_Float16*)carve((size_t)B * Ns * D * 2);
  _Float16* xg_h    = (_Float16*)carve((size_t)B * Ng * D * 2);
  _Float16* xg_hT   = (_Float16*)carve((size_t)B * Ng * D * 2);
  _Float16* xs_msg  = (_Float16*)carve((size_t)B * Ns * D * 2);
  _Float16* xg_msg  = (_Float16*)carve((size_t)B * Ng * D * 2);
  float*    rsum    = (float*)   carve((size_t)B * Ns * 4);

  k_prep<<<dim3(384), dim3(256), 0, stream>>>(Ws_pre, Wg_pre, Ws_post, Wg_post,
                                              WsPreT, WgPreT, WsPostT, WgPostT);
  k_pre<<<dim3((B * Ns) / 128), dim3(256), 0, stream>>>(xs, WsPreT, bs_pre, xs_h, xs_hT, Ns);
  k_pre<<<dim3((B * Ng) / 128), dim3(256), 0, stream>>>(xg, WgPreT, bg_pre, xg_h, xg_hT, Ng);
  k_msg_s<<<dim3((B * Ns) / 64), dim3(128), 0, stream>>>(vs, vg, xg_hT, xs_msg, rsum, Ns, Ng);
  k_msg_g<<<dim3((B * Ng) / 64), dim3(128), 0, stream>>>(vs, vg, xs_hT, rsum, xg_msg, Ns, Ng);
  k_post<<<dim3((B * Ns) / 128), dim3(256), 0, stream>>>(xs_h, xs_msg, WsPostT, bs_post,
                                                         out, Ns, 0, Ns + Ng);
  k_post<<<dim3((B * Ng) / 128), dim3(256), 0, stream>>>(xg_h, xg_msg, WgPostT, bg_post,
                                                         out, Ng, Ns, Ns + Ng);
}
